// GCN_1580547964985
// MI455X (gfx1250) — matrix operand order repaired and checked by reading, not compiled, not run on hardware
//
#include <hip/hip_runtime.h>

typedef __attribute__((ext_vector_type(16))) __bf16 v16bf;
typedef __attribute__((ext_vector_type(8)))  float  v8f;

#define MODE_F32_NOBIAS 0
#define MODE_BF16_RELU  1

__device__ __forceinline__ unsigned short f32_to_bf16(float f) {
    unsigned u = __float_as_uint(f);
    unsigned r = u + 0x7FFFu + ((u >> 16) & 1u);   // round-to-nearest-even
    return (unsigned short)(r >> 16);
}
__device__ __forceinline__ float bf16_to_f32(unsigned short h) {
    return __uint_as_float(((unsigned)h) << 16);
}

// ---------------- utility kernels ----------------

__global__ void fill_f32(float* p, int n, float v) {
    int i = blockIdx.x * blockDim.x + threadIdx.x;
    if (i < n) p[i] = v;
}

// W [K,Nc] fp32 row-major  ->  Wt [Nc,K] bf16 (so B-fragment k-pairs are contiguous)
__global__ void repack_w(const float* __restrict__ W, unsigned short* __restrict__ Wt,
                         int K, int Nc) {
    int idx = blockIdx.x * blockDim.x + threadIdx.x;
    if (idx >= K * Nc) return;
    int k = idx / Nc, n = idx % Nc;
    Wt[(size_t)n * K + k] = f32_to_bf16(W[idx]);
}

// ---------------- batchnorm ----------------

__global__ void bn_stats(const float* __restrict__ x, float* __restrict__ sum,
                         float* __restrict__ sq, int n) {
    int f = threadIdx.x & 63;
    int g = threadIdx.x >> 6;                 // 0..3 row groups per block
    float s = 0.f, s2 = 0.f;
    for (int r = blockIdx.x * 4 + g; r < n; r += gridDim.x * 4) {
        float v = x[(size_t)r * 64 + f];
        s += v; s2 += v * v;
    }
    atomicAdd(&sum[f], s);
    atomicAdd(&sq[f],  s2);
}

__global__ void bn_apply(const float* __restrict__ x, const float* __restrict__ sum,
                         const float* __restrict__ sq, const float* __restrict__ gamma,
                         const float* __restrict__ beta, unsigned short* __restrict__ out,
                         int n) {
    int idx = blockIdx.x * blockDim.x + threadIdx.x;
    if (idx >= n * 64) return;
    int f = idx & 63;
    const float invn = 1.0f / 50000.0f;
    float mu  = sum[f] * invn;
    float var = sq[f] * invn - mu * mu;
    float rs  = rsqrtf(var + 1e-5f);
    out[idx]  = f32_to_bf16((x[idx] - mu) * rs * gamma[f] + beta[f]);
}

// ---------------- WMMA GEMM: out = act(A[M,K] @ W[K,Nc] + b) ----------------
// A: bf16 row-major.  Bt: bf16 [Nc,K] (transposed weights).
// Block = 256 threads = 8 waves; waves tiled 4(M) x 2(N) -> 64x32 per block.
__global__ void gemm_wmma_bf16(const unsigned short* __restrict__ A,
                               const unsigned short* __restrict__ Bt,
                               const float* __restrict__ bias,
                               void* __restrict__ out,
                               int M, int K, int Nc, int mode) {
    int wave = threadIdx.x >> 5;
    int lane = threadIdx.x & 31;
    int row0 = blockIdx.y * 64 + (wave & 3) * 16;
    int col0 = blockIdx.x * 32 + (wave >> 2) * 16;
    if (row0 >= M) return;                     // uniform per-wave exit

    v8f c = {};
    int mA = row0 + (lane & 15);
    if (mA >= M) mA = M - 1;                   // clamp ragged tail loads
    const unsigned short* arow = A  + (size_t)mA * K;
    const unsigned short* brow = Bt + (size_t)(col0 + (lane & 15)) * K;
    const int aoff = (lane >> 4) << 3;         // A: lanes 16-31 take K+8 half
    const int boff = (lane >> 4) << 4;         // unused: B reads use ka, so lanes 16-31 take K+8 like A

    for (int k0 = 0; k0 < K; k0 += 32) {
        if (k0 + 32 < K) __builtin_prefetch(arow + k0 + 32, 0, 3);
        union { v16bf v; unsigned u[8]; } fa, fb;
#pragma unroll
        for (int r = 0; r < 8; ++r) {
            // A 16x32 bf16 fragment: k = half*16 + (lane>>4)*8 + pair*2
            int ka = k0 + aoff + ((r >> 2) << 4) + ((r & 3) << 1);
            fa.u[r] = *(const unsigned*)(arow + ka);
            // B 32x16 bf16 fragment: k = half*16 + (lane>>4)*8 + pair*2 (same as A)
            fb.u[r] = *(const unsigned*)(brow + ka);
        }
        c = __builtin_amdgcn_wmma_f32_16x16x32_bf16(false, fa.v, false, fb.v,
                                                    (short)0, c, false, false);
    }

    int n = col0 + (lane & 15);
    int mbase = row0 + ((lane >> 4) << 3);     // C/D: vgpr i -> M = i (+8 for hi lanes)
    float bv = bias ? bias[n] : 0.0f;
#pragma unroll
    for (int i = 0; i < 8; ++i) {
        int m = mbase + i;
        if (m < M) {
            float x = c[i] + bv;
            if (mode == MODE_BF16_RELU)
                ((unsigned short*)out)[(size_t)m * Nc + n] = f32_to_bf16(fmaxf(x, 0.0f));
            else
                ((float*)out)[(size_t)m * Nc + n] = x;
        }
    }
}

// ---------------- GCN conv pieces ----------------

__global__ void deg_count(const int* __restrict__ dst, float* __restrict__ deg, int e) {
    int i = blockIdx.x * blockDim.x + threadIdx.x;
    if (i < e) atomicAdd(&deg[dst[i]], 1.0f);
}

__global__ void rsqrt_inplace(float* __restrict__ d, int n) {
    int i = blockIdx.x * blockDim.x + threadIdx.x;
    if (i < n) d[i] = rsqrtf(d[i]);            // deg >= 1 always (self-loops)
}

// self-loop contribution: acc[i,f] = xw[i,f] * dinv[i]^2
__global__ void self_init(const float* __restrict__ xw, const float* __restrict__ dinv,
                          float* __restrict__ acc, int n) {
    int idx = blockIdx.x * blockDim.x + threadIdx.x;
    if (idx >= n * 128) return;
    float di = dinv[idx >> 7];
    acc[idx] = xw[idx] * di * di;
}

// one thread per (edge, 4-float chunk): 32 threads per edge
__global__ void edge_scatter(const int* __restrict__ src, const int* __restrict__ dst,
                             const float* __restrict__ dinv, const float* __restrict__ xw,
                             float* __restrict__ acc, int e) {
    int idx = blockIdx.x * blockDim.x + threadIdx.x;
    int ed = idx >> 5, cch = idx & 31;
    if (ed >= e) return;
    int s = src[ed], d = dst[ed];
    float w = dinv[s] * dinv[d];
    float4 v = ((const float4*)(xw + (size_t)s * 128))[cch];
    float* a = acc + (size_t)d * 128 + cch * 4;
    atomicAdd(a + 0, v.x * w);
    atomicAdd(a + 1, v.y * w);
    atomicAdd(a + 2, v.z * w);
    atomicAdd(a + 3, v.w * w);
}

// out = relu(acc + bias); write bf16 (mid convs) or fp32 (last conv, feeds pooling)
__global__ void bias_relu(const float* __restrict__ acc, const float* __restrict__ bias,
                          unsigned short* __restrict__ out_bf, float* __restrict__ out_f,
                          int n) {
    int idx = blockIdx.x * blockDim.x + threadIdx.x;
    if (idx >= n * 128) return;
    float v = fmaxf(acc[idx] + bias[idx & 127], 0.0f);
    if (out_bf) out_bf[idx] = f32_to_bf16(v);
    else        out_f[idx]  = v;
}

// ---------------- pooling + head ----------------

__global__ void pool_sum(const float* __restrict__ h, const int* __restrict__ batch,
                         float* __restrict__ psum, int n) {
    int idx = blockIdx.x * blockDim.x + threadIdx.x;
    if (idx >= n * 128) return;
    atomicAdd(&psum[(size_t)batch[idx >> 7] * 128 + (idx & 127)], h[idx]);
}

__global__ void pool_cnt(const int* __restrict__ batch, float* __restrict__ cnt, int n) {
    int i = blockIdx.x * blockDim.x + threadIdx.x;
    if (i < n) atomicAdd(&cnt[batch[i]], 1.0f);
}

__global__ void pool_fin(const float* __restrict__ psum, const float* __restrict__ cnt,
                         unsigned short* __restrict__ pcat, int towerIdx) {
    int idx = blockIdx.x * blockDim.x + threadIdx.x;
    if (idx >= 512 * 128) return;
    int b = idx >> 7, f = idx & 127;
    float v = psum[idx] / fmaxf(cnt[b], 1.0f);
    pcat[(size_t)b * 256 + towerIdx * 128 + f] = f32_to_bf16(v);
}

__global__ void head_dot(const unsigned short* __restrict__ h, const float* __restrict__ W5,
                         const float* __restrict__ b5, float* __restrict__ out) {
    int b = blockIdx.x * blockDim.x + threadIdx.x;
    if (b >= 512) return;
    float s = b5[0];
#pragma unroll 4
    for (int f = 0; f < 128; ++f) s += bf16_to_f32(h[(size_t)b * 128 + f]) * W5[f];
    out[b] = s;
}

// ---------------- host orchestration ----------------

extern "C" void kernel_launch(void* const* d_in, const int* in_sizes, int n_in,
                              void* d_out, int out_size, void* d_ws, size_t ws_size,
                              hipStream_t stream) {
    const int N = 50000, E = 600000;
    const float* xs[2]    = { (const float*)d_in[0], (const float*)d_in[3] };
    const int*   eis[2]   = { (const int*)d_in[1],   (const int*)d_in[4] };
    const int*   bats[2]  = { (const int*)d_in[2],   (const int*)d_in[5] };
    const float* gamma = (const float*)d_in[6];
    const float* beta  = (const float*)d_in[7];
    const float* W1 = (const float*)d_in[8];   const float* b1 = (const float*)d_in[9];
    const float* W2 = (const float*)d_in[10];  const float* b2 = (const float*)d_in[11];
    const float* Wc[3] = { (const float*)d_in[12], (const float*)d_in[14], (const float*)d_in[16] };
    const float* bc[3] = { (const float*)d_in[13], (const float*)d_in[15], (const float*)d_in[17] };
    const float* W3 = (const float*)d_in[18];  const float* b3 = (const float*)d_in[19];
    const float* W4 = (const float*)d_in[20];  const float* b4 = (const float*)d_in[21];
    const float* W5 = (const float*)d_in[22];  const float* b5 = (const float*)d_in[23];

    // carve workspace
    char* p = (char*)d_ws;
    auto carve = [&](size_t bytes) { void* r = (void*)p; p += (bytes + 255) & ~(size_t)255; return r; };
    unsigned short* W1t  = (unsigned short*)carve((size_t)256 * 64  * 2);
    unsigned short* W2t  = (unsigned short*)carve((size_t)128 * 256 * 2);
    unsigned short* Wct[3];
    for (int i = 0; i < 3; ++i) Wct[i] = (unsigned short*)carve((size_t)128 * 128 * 2);
    unsigned short* W3t  = (unsigned short*)carve((size_t)256 * 256 * 2);
    unsigned short* W4t  = (unsigned short*)carve((size_t)128 * 256 * 2);
    float* bnsum = (float*)carve(64 * 4);
    float* bnsq  = (float*)carve(64 * 4);
    unsigned short* xbn = (unsigned short*)carve((size_t)N * 64  * 2);
    unsigned short* h1  = (unsigned short*)carve((size_t)N * 256 * 2);
    unsigned short* hA  = (unsigned short*)carve((size_t)N * 128 * 2);
    unsigned short* hB  = h1;                  // reuse: h1 dead after GEMM2
    float* xw   = (float*)carve((size_t)N * 128 * 4);
    float* acc  = (float*)carve((size_t)N * 128 * 4);
    float* dinv = (float*)carve((size_t)N * 4);
    float* psum = (float*)carve((size_t)512 * 128 * 4);
    float* pcnt = (float*)carve((size_t)512 * 4);
    unsigned short* pcat = (unsigned short*)carve((size_t)512 * 256 * 2);
    unsigned short* hh1  = (unsigned short*)carve((size_t)512 * 256 * 2);
    unsigned short* hh2  = (unsigned short*)carve((size_t)512 * 128 * 2);
    float* h3f = xw;                           // reuse xw for last conv output

    dim3 blk(256);
    auto cdiv = [](int a, int b) { return (a + b - 1) / b; };

    // repack all weights into WMMA-B order (bf16, [Nc,K])
    repack_w<<<cdiv(64 * 256, 256), blk, 0, stream>>>(W1, W1t, 64, 256);
    repack_w<<<cdiv(256 * 128, 256), blk, 0, stream>>>(W2, W2t, 256, 128);
    for (int i = 0; i < 3; ++i)
        repack_w<<<cdiv(128 * 128, 256), blk, 0, stream>>>(Wc[i], Wct[i], 128, 128);
    repack_w<<<cdiv(256 * 256, 256), blk, 0, stream>>>(W3, W3t, 256, 256);
    repack_w<<<cdiv(256 * 128, 256), blk, 0, stream>>>(W4, W4t, 256, 128);

    for (int t = 0; t < 2; ++t) {
        const float* x = xs[t];
        const int* src = eis[t];        // edge_index row 0
        const int* dst = eis[t] + E;    // edge_index row 1
        const int* bat = bats[t];

        // batchnorm
        fill_f32<<<1, 64, 0, stream>>>(bnsum, 64, 0.0f);
        fill_f32<<<1, 64, 0, stream>>>(bnsq, 64, 0.0f);
        bn_stats<<<128, blk, 0, stream>>>(x, bnsum, bnsq, N);
        bn_apply<<<cdiv(N * 64, 256), blk, 0, stream>>>(x, bnsum, bnsq, gamma, beta, xbn, N);

        // dense layers (WMMA bf16)
        gemm_wmma_bf16<<<dim3(256 / 32, cdiv(N, 64)), blk, 0, stream>>>(
            xbn, W1t, b1, h1, N, 64, 256, MODE_BF16_RELU);
        gemm_wmma_bf16<<<dim3(128 / 32, cdiv(N, 64)), blk, 0, stream>>>(
            h1, W2t, b2, hA, N, 256, 128, MODE_BF16_RELU);

        // symmetric-norm degrees (self-loop => init 1)
        fill_f32<<<cdiv(N, 256), blk, 0, stream>>>(dinv, N, 1.0f);
        deg_count<<<cdiv(E, 256), blk, 0, stream>>>(dst, dinv, E);
        rsqrt_inplace<<<cdiv(N, 256), blk, 0, stream>>>(dinv, N);

        // 3x GCNConv
        const unsigned short* hin = hA;
        unsigned short* hout = hB;
        for (int c = 0; c < 3; ++c) {
            gemm_wmma_bf16<<<dim3(128 / 32, cdiv(N, 64)), blk, 0, stream>>>(
                hin, Wct[c], nullptr, xw, N, 128, 128, MODE_F32_NOBIAS);
            self_init<<<cdiv(N * 128, 256), blk, 0, stream>>>(xw, dinv, acc, N);
            edge_scatter<<<cdiv(E * 32, 256), blk, 0, stream>>>(src, dst, dinv, xw, acc, E);
            if (c < 2) {
                bias_relu<<<cdiv(N * 128, 256), blk, 0, stream>>>(acc, bc[c], hout, nullptr, N);
                const unsigned short* tmp = hin; hin = hout; hout = (unsigned short*)tmp;
            } else {
                bias_relu<<<cdiv(N * 128, 256), blk, 0, stream>>>(acc, bc[c], nullptr, h3f, N);
            }
        }

        // mean pool per graph
        fill_f32<<<cdiv(512 * 128, 256), blk, 0, stream>>>(psum, 512 * 128, 0.0f);
        fill_f32<<<cdiv(512, 256), blk, 0, stream>>>(pcnt, 512, 0.0f);
        pool_sum<<<cdiv(N * 128, 256), blk, 0, stream>>>(h3f, bat, psum, N);
        pool_cnt<<<cdiv(N, 256), blk, 0, stream>>>(bat, pcnt, N);
        pool_fin<<<cdiv(512 * 128, 256), blk, 0, stream>>>(psum, pcnt, pcat, t);
    }

    // head MLP
    gemm_wmma_bf16<<<dim3(256 / 32, cdiv(512, 64)), blk, 0, stream>>>(
        pcat, W3t, b3, hh1, 512, 256, 256, MODE_BF16_RELU);
    gemm_wmma_bf16<<<dim3(128 / 32, cdiv(512, 64)), blk, 0, stream>>>(
        hh1, W4t, b4, hh2, 512, 256, 128, MODE_BF16_RELU);
    head_dot<<<cdiv(512, 256), blk, 0, stream>>>(hh2, W5, b5, (float*)d_out);
}
